// FusedMLP5Softplus_69286412419483
// MI455X (gfx1250) — matrix operand order repaired and checked by reading, not compiled, not run on hardware
//
#include <hip/hip_runtime.h>
#include <hip/hip_bf16.h>

typedef __attribute__((ext_vector_type(16))) __bf16 v16bf;
typedef __attribute__((ext_vector_type(8)))  __bf16 v8bf;
typedef __attribute__((ext_vector_type(4)))  __bf16 v4bf;
typedef __attribute__((ext_vector_type(8)))  float  v8f;

#define DDIM   1024
#define MROWS  32768
#define BM     64
#define LDSW   (DDIM + 8)   // padded LDS row stride (bf16 elements), keeps 16B row alignment
#define NLAYER 5
// Packed-B geometry: a tile is 16 cols x 32 K = 512 bf16 (1KB), lane-fragment order.
// Tiles ordered: coltile c (0..63) major, K-step k (0..31) minor -> tile id = c*32 + k.
#define TILE_ELEMS 512
#define KSTEPS     (DDIM / 32)          // 32
#define CTILES     (DDIM / 16)          // 64

__device__ __forceinline__ float softplus_f(float x) {
    // max(x,0) + log(1 + exp(-|x|)) using hardware exp/log; error << bf16 ulp
    return fmaxf(x, 0.0f) + __logf(1.0f + __expf(-fabsf(x)));
}

// ---------------------------------------------------------------------------
// Prep: pack weights into WMMA B-fragment order, fp32 -> bf16.
// For layer l, coltile c, K-step k, lane L (m16=L&15, kh=L>>4), element e:
//   wPack[((l*2048 + c*32 + k) * 512) + L*16 + e] = bf16( w[(k*32 + kh*16 + e)*D + c*16 + m16] )
// ---------------------------------------------------------------------------
__global__ void wt_pack_kernel(const float* __restrict__ w1,
                               const float* __restrict__ w2,
                               const float* __restrict__ w3,
                               const float* __restrict__ w4,
                               const float* __restrict__ w5,
                               __bf16* __restrict__ wPack) {
    __shared__ float tile[32][17];
    const int l = blockIdx.y;
    const float* w = (l == 0) ? w1 : (l == 1) ? w2 : (l == 2) ? w3 : (l == 3) ? w4 : w5;
    const int bx = blockIdx.x;          // tile id = c*32 + k
    const int c  = bx >> 5;
    const int k  = bx & 31;
    const int t  = threadIdx.x;         // 0..127

    #pragma unroll
    for (int i = t; i < 512; i += 128) {
        const int row = i >> 4;
        const int col = i & 15;
        tile[row][col] = w[(size_t)(k * 32 + row) * DDIM + c * 16 + col];
    }
    __syncthreads();

    const int d0   = t * 4;
    const int lane = d0 >> 4;
    const int e0   = d0 & 15;
    const int kh   = lane >> 4;
    const int m16  = lane & 15;
    v4bf v;
    #pragma unroll
    for (int j = 0; j < 4; ++j)
        v[j] = (__bf16)tile[kh * 16 + e0 + j][m16];
    __bf16* dst = wPack + ((size_t)l * (CTILES * KSTEPS) + bx) * TILE_ELEMS + d0;
    *(v4bf*)dst = v;
}

// ---------------------------------------------------------------------------
// Fused 5-layer MLP. One block = 64 rows. 256 threads = 8 waves (wave32).
// Wave tile: 64 rows x 32 cols = 8 accumulators (4 strips x 2 col-tiles).
// B fragments (global/L2) double-buffered one K-step ahead; A fragments
// (LDS, short latency) single-buffered to stay under 256 VGPRs.
// ---------------------------------------------------------------------------
__global__ __launch_bounds__(256, 1)
void mlp5_fused_kernel(const float* __restrict__ x,
                       const __bf16* __restrict__ wPack,  // 5 * 2048 tiles * 512
                       float* __restrict__ out) {
    __shared__ __align__(16) __bf16 hbufA[BM * LDSW];
    __shared__ __align__(16) __bf16 hbufB[BM * LDSW];

    const int tid  = threadIdx.x;
    const int wave = tid >> 5;
    const int lane = tid & 31;
    const int m16  = lane & 15;
    const int kh   = lane >> 4;        // which 16-lane half
    const size_t rowBase = (size_t)blockIdx.x * BM;

    // ---- load x row-tile, fp32 -> bf16 into hbufA ----
    {
        const float4* xv = (const float4*)(x + rowBase * DDIM);
        for (int idx = tid; idx < BM * DDIM / 4; idx += 256) {
            const int r = idx >> 8;              // idx / (D/4)
            const int c = (idx & 255) << 2;      // (idx % (D/4)) * 4
            float4 v = xv[idx];
            __bf16* p = &hbufA[r * LDSW + c];
            p[0] = (__bf16)v.x; p[1] = (__bf16)v.y;
            p[2] = (__bf16)v.z; p[3] = (__bf16)v.w;
        }
    }
    __syncthreads();

    __bf16* hin  = hbufA;
    __bf16* hout = hbufB;

    for (int layer = 0; layer < NLAYER; ++layer) {
        const __bf16* wl = wPack + (size_t)layer * (CTILES * KSTEPS) * TILE_ELEMS;

        for (int chunk = 0; chunk < 4; ++chunk) {
            const int cbase = chunk * 256 + wave * 32;   // this wave's 32 output cols
            const int c0    = cbase >> 4;                // first of 2 col-tiles
            v8f acc[4][2] = {{v8f{}, v8f{}}, {v8f{}, v8f{}},
                             {v8f{}, v8f{}}, {v8f{}, v8f{}}};

            // packed-B pointers: contiguous 1KB per K-step, coalesced across lanes
            const __bf16* wp0 = wl + (size_t)c0 * KSTEPS * TILE_ELEMS + lane * 16;
            const __bf16* wp1 = wp0 + (size_t)KSTEPS * TILE_ELEMS;

            // B double-buffered (global latency), A single-buffered (LDS latency)
            v16bf bfr[2][2];
            v16bf afr[4];

            auto loadB = [&](int ks, int buf) {
                const v8bf* bp0 = (const v8bf*)(wp0 + ks * TILE_ELEMS);
                const v8bf* bp1 = (const v8bf*)(wp1 + ks * TILE_ELEMS);
                v8bf b00 = bp0[0], b01 = bp0[1];
                v8bf b10 = bp1[0], b11 = bp1[1];
                #pragma unroll
                for (int e = 0; e < 8; ++e) {
                    bfr[buf][0][e] = b00[e]; bfr[buf][0][e + 8] = b01[e];
                    bfr[buf][1][e] = b10[e]; bfr[buf][1][e + 8] = b11[e];
                }
            };
            auto loadA = [&](int ks) {
                const int kb = ks * 32;
                #pragma unroll
                for (int s = 0; s < 4; ++s) {
                    const __bf16* ar = hin + (s * 16 + m16) * LDSW + kb + kh * 16;
                    v8bf a0 = *(const v8bf*)(ar);
                    v8bf a1 = *(const v8bf*)(ar + 8);
                    #pragma unroll
                    for (int e = 0; e < 8; ++e) {
                        afr[s][e] = a0[e]; afr[s][e + 8] = a1[e];
                    }
                }
            };
            auto compute = [&](int buf) {
                #pragma unroll
                for (int s = 0; s < 4; ++s) {
                    acc[s][0] = __builtin_amdgcn_wmma_f32_16x16x32_bf16(
                        false, afr[s], false, bfr[buf][0], (short)0,
                        acc[s][0], false, false);
                    acc[s][1] = __builtin_amdgcn_wmma_f32_16x16x32_bf16(
                        false, afr[s], false, bfr[buf][1], (short)0,
                        acc[s][1], false, false);
                }
            };

            // prologue
            loadB(0, 0);
            loadA(0);
            // steady state: B for ks+1 issued before compute on ks; A reloaded
            // right after the compute block that last reads it
            for (int ks = 0; ks < KSTEPS - 2; ks += 2) {
                loadB(ks + 1, 1);
                compute(0);
                loadA(ks + 1);
                loadB(ks + 2, 0);
                compute(1);
                loadA(ks + 2);
            }
            loadB(KSTEPS - 1, 1);
            compute(0);
            loadA(KSTEPS - 1);
            compute(1);

            // ---- epilogue: C/D layout row = v + 8*kh, col = m16 ----
            if (layer < NLAYER - 1) {
                #pragma unroll
                for (int s = 0; s < 4; ++s) {
                    #pragma unroll
                    for (int t = 0; t < 2; ++t) {
                        #pragma unroll
                        for (int v = 0; v < 8; ++v) {
                            const int r = s * 16 + v + kh * 8;
                            hout[r * LDSW + cbase + t * 16 + m16] =
                                (__bf16)softplus_f(acc[s][t][v]);
                        }
                    }
                }
            } else {
                float* orow = out + rowBase * DDIM;
                #pragma unroll
                for (int s = 0; s < 4; ++s) {
                    #pragma unroll
                    for (int t = 0; t < 2; ++t) {
                        #pragma unroll
                        for (int v = 0; v < 8; ++v) {
                            const int r = s * 16 + v + kh * 8;
                            orow[(size_t)r * DDIM + cbase + t * 16 + m16] = acc[s][t][v];
                        }
                    }
                }
            }
        }
        __syncthreads();                 // layer output complete
        __bf16* t = hin; hin = hout; hout = t;
    }
}

extern "C" void kernel_launch(void* const* d_in, const int* in_sizes, int n_in,
                              void* d_out, int out_size, void* d_ws, size_t ws_size,
                              hipStream_t stream) {
    const float* x  = (const float*)d_in[0];
    const float* w1 = (const float*)d_in[1];
    const float* w2 = (const float*)d_in[2];
    const float* w3 = (const float*)d_in[3];
    const float* w4 = (const float*)d_in[4];
    const float* w5 = (const float*)d_in[5];
    float* out = (float*)d_out;

    __bf16* wPack = (__bf16*)d_ws;   // 5 * 1024 * 1024 * 2B = 10 MB scratch

    dim3 pThreads(128);
    dim3 pGrid(CTILES * KSTEPS, 5);
    wt_pack_kernel<<<pGrid, pThreads, 0, stream>>>(w1, w2, w3, w4, w5, wPack);

    mlp5_fused_kernel<<<MROWS / BM, 256, 0, stream>>>(x, wPack, out);
}
